// CNOFactorized2DBlock_65085934403665
// MI455X (gfx1250) — matrix operand order repaired and checked by reading, not compiled, not run on hardware
//
#include <hip/hip_runtime.h>
#include <hip/hip_bf16.h>

// ---------------------------------------------------------------------------
// CNO factorized 2D block, CDNA5 (gfx1250) WMMA implementation.
// All GEMMs run through v_wmma_f32_16x16x32_f16 (f16 in, f32 accumulate).
// Pure-f16 LDS staging uses global_load_async_to_lds_b128 (ASYNCcnt path).
// B=4, M=N=256, C=64, MODES=32, HID=128, N_LAYERS=4.
// ---------------------------------------------------------------------------

typedef __attribute__((ext_vector_type(16))) _Float16 v16h;
typedef __attribute__((ext_vector_type(8)))  _Float16 v8h;
typedef __attribute__((ext_vector_type(8)))  float    v8f;

#define WMMA_F16(a, b, c) \
  __builtin_amdgcn_wmma_f32_16x16x32_f16(false, (a), false, (b), (short)0, (c), false, false)

// Async copy of 16 bytes global -> LDS. Generic pointers to __shared__ carry
// the LDS byte offset in their low 32 bits (shared-aperture encoding), which
// is what the instruction's VDST operand wants (the VGPR gives the LDS byte address).
__device__ __forceinline__ void async_copy_b128(void* lds_dst, const void* gsrc) {
  asm volatile("global_load_async_to_lds_b128 %0, %1, off"
               :
               : "v"((unsigned)(unsigned long long)lds_dst),
                 "v"((unsigned long long)gsrc)
               : "memory");
}
__device__ __forceinline__ void async_wait0() {
  asm volatile("s_wait_asynccnt 0x0" ::: "memory");
}

// A fragment (16x32 f16, MxK): lane&15 = row M; lanes 0-15 hold K 0-7 & 16-23,
// lanes 16-31 hold K 8-15 & 24-31 (ISA 7.12.2). LDS tile is [row][k], row
// stride in halves must be a multiple of 8 (16B alignment).
__device__ __forceinline__ v16h frag_a(const _Float16* base, int stride, int lane) {
  const _Float16* p = base + (lane & 15) * stride + ((lane & 16) ? 8 : 0);
  v8h lo = *(const v8h*)(p);
  v8h hi = *(const v8h*)(p + 16);
  v16h r;
#pragma unroll
  for (int i = 0; i < 8; ++i) { r[i] = lo[i]; r[i + 8] = hi[i]; }
  return r;
}

// B fragment (32x16 f16, KxN): lane&15 = col N; lanes 0-15 hold K 0-7 & 16-23,
// lanes 16-31 hold K 8-15 & 24-31. LDS tile is stored transposed: [n][k], row
// stride multiple of 8 halves (16B alignment).
__device__ __forceinline__ v16h frag_b(const _Float16* base, int stride, int lane) {
  const _Float16* p = base + (lane & 15) * stride + ((lane & 16) ? 8 : 0);
  v8h lo = *(const v8h*)(p);
  v8h hi = *(const v8h*)(p + 16);
  v16h r;
#pragma unroll
  for (int i = 0; i < 8; ++i) { r[i] = lo[i]; r[i + 8] = hi[i]; }
  return r;
}

// C/D 16x16 f32: VGPR v holds row v (lanes 0-15) or v+8 (lanes 16-31), col = lane&15.

// ---------------------------------------------------------------------------
// DCT matrix fill: D16[32][256] (truncated forward) and Dt16[256][32] (inverse).
__global__ __launch_bounds__(256) void k_fill_dct(_Float16* D16, _Float16* Dt16) {
  int n = threadIdx.x;  // 0..255
  const float s = 0.0883883476483184f;  // sqrt(2/256)
  for (int k = 0; k < 32; ++k) {
    float v = __cosf(3.14159265358979323f * (n + 0.5f) * (float)k * (1.0f / 256.0f)) * s;
    if (k == 0) v *= 0.70710678118654752f;
    D16[k * 256 + n]  = (_Float16)v;
    Dt16[n * 32 + k]  = (_Float16)v;
  }
}

// ---------------------------------------------------------------------------
// Lift: xb[p][c] = sum_j x[p][j] * W_in[c][j] + b_in[c]   (K=12, memory bound)
__global__ __launch_bounds__(256) void k_lift(const float* __restrict__ x,
                                              const float* __restrict__ W_in,
                                              const float* __restrict__ b_in,
                                              float* __restrict__ xb) {
  int t = blockIdx.x * 256 + threadIdx.x;  // over 262144*64
  int c = t & 63, p = t >> 6;
  const float* xr = x + p * 12;
  const float* wr = W_in + c * 12;
  float acc = b_in[c];
#pragma unroll
  for (int j = 0; j < 12; ++j) acc += xr[j] * wr[j];
  xb[t] = acc;
}

// ---------------------------------------------------------------------------
// Forward DCT: for slice p (b = p>>8, r = p&255), base = b*4194304 + r*RS,
// row stride RO. Computes Xf[km][p][i] = sum_n D[km][n] * src[base + n*RO + i].
// C = 32x64, K = 256.
#define KSTF (256 + 16)
__global__ __launch_bounds__(256) void k_dct_fwd(const float* __restrict__ src,
                                                 const _Float16* __restrict__ D16,
                                                 _Float16* __restrict__ Xf,
                                                 int RS, int RO) {
  __shared__ _Float16 lA[32 * KSTF];  // D tile  [km][n]
  __shared__ _Float16 lB[64 * KSTF];  // X tile  [i][n] (transposed)
  int t = threadIdx.x;
  int p = blockIdx.x;
  int base = (p >> 8) * 4194304 + (p & 255) * RS;

  // D16 is f16 already: async straight copy, 32 rows x 32 b128-chunks.
  for (int idx = t; idx < 1024; idx += 256) {
    int r = idx >> 5, c = idx & 31;
    async_copy_b128(&lA[r * KSTF + c * 8], &D16[r * 256 + c * 8]);
  }
  // X tile needs f32->f16 + transpose: VALU path.
  int tx = t & 63, ty = t >> 6;
  for (int n = ty; n < 256; n += 4)
    lB[tx * KSTF + n] = (_Float16)src[base + n * RO + tx];
  async_wait0();
  __syncthreads();

  int wave = t >> 5, lane = t & 31;
  int mt = wave >> 2, nt = wave & 3;  // 2 x 4 tiles, one per wave
  v8f acc = {};
#pragma unroll
  for (int kk = 0; kk < 8; ++kk) {
    v16h a = frag_a(&lA[(mt * 16) * KSTF + kk * 32], KSTF, lane);
    v16h b = frag_b(&lB[(nt * 16) * KSTF + kk * 32], KSTF, lane);
    acc = WMMA_F16(a, b, acc);
  }
  int col = lane & 15, rb = (lane & 16) ? 8 : 0;
#pragma unroll
  for (int v = 0; v < 8; ++v) {
    int km = mt * 16 + rb + v;
    int i  = nt * 16 + col;
    Xf[((km << 10) + p) * 64 + i] = (_Float16)acc[v];
  }
}

// ---------------------------------------------------------------------------
// Per-mode channel mix: for mode k, rows p0..p0+63:
//   Ym[p][o][k] = sum_i Xf[k][p][i] * w[(i*64+o)*32 + k]
// (note transposed [o][k] output layout so the inverse-DCT stage can
//  async-copy its B tile without transposition)
// C = 64x64, K = 64.
__global__ __launch_bounds__(256) void k_mode_mix(const _Float16* __restrict__ Xf,
                                                  const float* __restrict__ w,
                                                  _Float16* __restrict__ Ym) {
  __shared__ _Float16 lA[64 * 80];  // Xf tile [row][i]
  __shared__ _Float16 lB[64 * 80];  // W tile  [o][i] (transposed)
  int t = threadIdx.x;
  int k = blockIdx.x, p0 = blockIdx.y * 64;
  __builtin_prefetch(w + k, 0, 0);
  // Xf tile is f16: async straight copy, 64 rows x 8 b128-chunks.
  for (int idx = t; idx < 512; idx += 256) {
    int r = idx >> 3, c = idx & 7;
    async_copy_b128(&lA[r * 80 + c * 8], &Xf[((k << 10) + p0 + r) * 64 + c * 8]);
  }
  // Weight gather (f32, stride 32) + convert: VALU path.
  for (int idx = t; idx < 64 * 64; idx += 256) {
    int r = idx >> 6, c = idx & 63;
    lB[c * 80 + r] = (_Float16)w[(r * 64 + c) * 32 + k];  // r=i, c=o
  }
  async_wait0();
  __syncthreads();

  int wave = t >> 5, lane = t & 31;
  int col = lane & 15, rb = (lane & 16) ? 8 : 0;
#pragma unroll
  for (int s = 0; s < 2; ++s) {
    int tt = wave * 2 + s, mt = tt >> 2, nt = tt & 3;
    v8f acc = {};
#pragma unroll
    for (int kk = 0; kk < 2; ++kk) {
      v16h a = frag_a(&lA[(mt * 16) * 80 + kk * 32], 80, lane);
      v16h b = frag_b(&lB[(nt * 16) * 80 + kk * 32], 80, lane);
      acc = WMMA_F16(a, b, acc);
    }
#pragma unroll
    for (int v = 0; v < 8; ++v) {
      int p = p0 + mt * 16 + rb + v;
      int o = nt * 16 + col;
      Ym[(p * 64 + o) * 32 + k] = (_Float16)acc[v];  // [p][o][k]
    }
  }
}

// ---------------------------------------------------------------------------
// Inverse DCT: dst[base + n*RO + o] (=|+=) sum_k Dt[n][k] * Ym[p][o][k]
// C = 64x64 per block (grid.y covers n in 4 chunks), K = 32.
__global__ __launch_bounds__(256) void k_dct_inv(const _Float16* __restrict__ Dt16,
                                                 const _Float16* __restrict__ Ym,
                                                 float* __restrict__ dst,
                                                 int RS, int RO, int accflag) {
  __shared__ _Float16 lA[64 * 48];  // Dt tile [n][k]
  __shared__ _Float16 lB[64 * 48];  // Ym tile [o][k]
  int t = threadIdx.x;
  int p = blockIdx.x, n0 = blockIdx.y * 64;
  int base = (p >> 8) * 4194304 + (p & 255) * RS;
  // Both tiles are f16 straight copies: 64 rows x 4 b128-chunks each.
  {
    int r = t >> 2, c = t & 3;  // one chunk per thread per tile
    async_copy_b128(&lA[r * 48 + c * 8], &Dt16[(n0 + r) * 32 + c * 8]);
    async_copy_b128(&lB[r * 48 + c * 8], &Ym[(p * 64 + r) * 32 + c * 8]);
  }
  async_wait0();
  __syncthreads();

  int wave = t >> 5, lane = t & 31;
  int col = lane & 15, rb = (lane & 16) ? 8 : 0;
#pragma unroll
  for (int s = 0; s < 2; ++s) {
    int tt = wave * 2 + s, mt = tt >> 2, nt = tt & 3;
    v16h a = frag_a(&lA[(mt * 16) * 48], 48, lane);
    v16h b = frag_b(&lB[(nt * 16) * 48], 48, lane);
    v8f acc = {};
    acc = WMMA_F16(a, b, acc);
#pragma unroll
    for (int v = 0; v < 8; ++v) {
      int n = n0 + mt * 16 + rb + v;
      int o = nt * 16 + col;
      int gi = base + n * RO + o;
      float r = acc[v];
      dst[gi] = accflag ? (dst[gi] + r) : r;
    }
  }
}

// ---------------------------------------------------------------------------
// Fused FFN + residual: t = relu(h*W1^T + b1); bb = t*W2^T + b2;
// bout = bb (overwrites h buffer), x += bb.   GEMM1: 64x128 K=64; GEMM2: 64x64 K=128.
__global__ __launch_bounds__(256) void k_ffn(const float* h,
                                             const float* __restrict__ W1,
                                             const float* __restrict__ b1,
                                             const float* __restrict__ W2,
                                             const float* __restrict__ b2,
                                             float* __restrict__ x,
                                             float* bout) {
  __shared__ _Float16 smem[64 * 80 + 128 * 80];  // lH + lW1, later reused for lW2
  __shared__ _Float16 lT[64 * 144];              // relu intermediate [row][o]
  _Float16* lH  = smem;            // [row][i], stride 80
  _Float16* lW1 = smem + 64 * 80;  // [o][i],  stride 80
  _Float16* lW2 = smem;            // phase 2: [o][j], stride 144 (9216 <= 15360)
  int t = threadIdx.x;
  int p0 = blockIdx.x * 64;
  __builtin_prefetch(W1, 0, 0);
  __builtin_prefetch(W2, 0, 0);

  for (int idx = t; idx < 64 * 64; idx += 256) {
    int r = idx >> 6, c = idx & 63;
    lH[r * 80 + c] = (_Float16)h[(p0 + r) * 64 + c];
  }
  for (int idx = t; idx < 128 * 64; idx += 256) {
    int o = idx >> 6, i = idx & 63;
    lW1[o * 80 + i] = (_Float16)W1[o * 64 + i];
  }
  __syncthreads();

  int wave = t >> 5, lane = t & 31;
  int col = lane & 15, rbs = (lane & 16) ? 8 : 0;
  // GEMM1: 4x8 tiles, 4 per wave
#pragma unroll
  for (int s = 0; s < 4; ++s) {
    int tt = wave * 4 + s, mt = tt >> 3, nt = tt & 7;
    v8f acc = {};
#pragma unroll
    for (int kk = 0; kk < 2; ++kk) {
      v16h a = frag_a(&lH[(mt * 16) * 80 + kk * 32], 80, lane);
      v16h b = frag_b(&lW1[(nt * 16) * 80 + kk * 32], 80, lane);
      acc = WMMA_F16(a, b, acc);
    }
#pragma unroll
    for (int v = 0; v < 8; ++v) {
      int row = mt * 16 + rbs + v, o = nt * 16 + col;
      float z = acc[v] + b1[o];
      lT[row * 144 + o] = (_Float16)(z > 0.0f ? z : 0.0f);
    }
  }
  __syncthreads();
  for (int idx = t; idx < 64 * 128; idx += 256) {
    int o = idx >> 7, j = idx & 127;
    lW2[o * 144 + j] = (_Float16)W2[o * 128 + j];
  }
  __syncthreads();
  // GEMM2: 4x4 tiles, 2 per wave
#pragma unroll
  for (int s = 0; s < 2; ++s) {
    int tt = wave * 2 + s, mt = tt >> 2, nt = tt & 3;
    v8f acc = {};
#pragma unroll
    for (int kk = 0; kk < 4; ++kk) {
      v16h a = frag_a(&lT[(mt * 16) * 144 + kk * 32], 144, lane);
      v16h b = frag_b(&lW2[(nt * 16) * 144 + kk * 32], 144, lane);
      acc = WMMA_F16(a, b, acc);
    }
#pragma unroll
    for (int v = 0; v < 8; ++v) {
      int row = mt * 16 + rbs + v, o = nt * 16 + col;
      int gi = (p0 + row) * 64 + o;
      float bb = acc[v] + b2[o];
      bout[gi] = bb;
      x[gi] += bb;
    }
  }
}

// ---------------------------------------------------------------------------
// Head fold: forecast = b @ (W_o2 @ W_o1)^T + (W_o2 @ b_o1 + b_o2)
__global__ void k_head_fold(const float* __restrict__ Wo1, const float* __restrict__ bo1,
                            const float* __restrict__ Wo2, const float* __restrict__ bo2,
                            float* __restrict__ wc) {
  int c = threadIdx.x;
  if (c < 64) {
    float a = 0.0f;
    for (int j = 0; j < 128; ++j) a += Wo2[j] * Wo1[j * 64 + c];
    wc[c] = a;
  } else if (c == 64) {
    float a = bo2[0];
    for (int j = 0; j < 128; ++j) a += Wo2[j] * bo1[j];
    wc[64] = a;
  }
}

__global__ __launch_bounds__(256) void k_head(const float* __restrict__ bsrc,
                                              const float* __restrict__ wc,
                                              float* __restrict__ out) {
  int p = blockIdx.x * 256 + threadIdx.x;
  float a = wc[64];
  const float* r = bsrc + p * 64;
#pragma unroll
  for (int c = 0; c < 64; ++c) a += r[c] * wc[c];
  out[p] = a;
}

// ---------------------------------------------------------------------------
extern "C" void kernel_launch(void* const* d_in, const int* in_sizes, int n_in,
                              void* d_out, int out_size, void* d_ws, size_t ws_size,
                              hipStream_t stream) {
  (void)in_sizes; (void)n_in; (void)out_size; (void)ws_size;
  const float* x    = (const float*)d_in[0];
  const float* W_in = (const float*)d_in[1];
  const float* b_in = (const float*)d_in[2];
  const float* fw0  = (const float*)d_in[3];
  const float* fw1  = (const float*)d_in[4];
  const float* ffW1 = (const float*)d_in[5];
  const float* ffb1 = (const float*)d_in[6];
  const float* ffW2 = (const float*)d_in[7];
  const float* ffb2 = (const float*)d_in[8];
  const float* W_o1 = (const float*)d_in[9];
  const float* b_o1 = (const float*)d_in[10];
  const float* W_o2 = (const float*)d_in[11];
  const float* b_o2 = (const float*)d_in[12];
  float* out = (float*)d_out;

  char* ws = (char*)d_ws;
  float*    xbuf = (float*)(ws);                         // 64 MB activation
  float*    hbuf = (float*)(ws + 67108864);              // 64 MB fourier/b buffer
  _Float16* Xf   = (_Float16*)(ws + 134217728);          // 4 MB mode-space fwd
  _Float16* Ym   = (_Float16*)(ws + 138412032);          // 4 MB mode-space mixed
  _Float16* D16  = (_Float16*)(ws + 142606336);          // 16 KB fwd DCT (32x256)
  _Float16* Dt16 = (_Float16*)(ws + 142606336 + 16384);  // 16 KB inv DCT (256x32)
  float*    wc   = (float*)(ws + 142606336 + 32768);     // folded head (65 f32)

  k_fill_dct<<<1, 256, 0, stream>>>(D16, Dt16);
  k_head_fold<<<1, 128, 0, stream>>>(W_o1, b_o1, W_o2, b_o2, wc);
  k_lift<<<65536, 256, 0, stream>>>(x, W_in, b_in, xbuf);

  for (int l = 0; l < 4; ++l) {
    const float* w0 = fw0 + l * 131072;
    const float* w1 = fw1 + l * 131072;
    // Y direction (DCT over n; slice stride RS=16384, row stride RO=64)
    k_dct_fwd<<<1024, 256, 0, stream>>>(xbuf, D16, Xf, 16384, 64);
    k_mode_mix<<<dim3(32, 16), 256, 0, stream>>>(Xf, w0, Ym);
    k_dct_inv<<<dim3(1024, 4), 256, 0, stream>>>(Dt16, Ym, hbuf, 16384, 64, 0);
    // X direction (DCT over m; slice stride RS=64, row stride RO=16384)
    k_dct_fwd<<<1024, 256, 0, stream>>>(xbuf, D16, Xf, 64, 16384);
    k_mode_mix<<<dim3(32, 16), 256, 0, stream>>>(Xf, w1, Ym);
    k_dct_inv<<<dim3(1024, 4), 256, 0, stream>>>(Dt16, Ym, hbuf, 64, 16384, 1);
    // FFN + residual; b (FF output) written back into hbuf for the head
    k_ffn<<<4096, 256, 0, stream>>>(hbuf, ffW1 + l * 8192, ffb1 + l * 128,
                                    ffW2 + l * 8192, ffb2 + l * 64, xbuf, hbuf);
  }
  k_head<<<1024, 256, 0, stream>>>(hbuf, wc, out);
}
